// Block_21646635172046
// MI455X (gfx1250) — matrix operand order repaired and checked by reading, not compiled, not run on hardware
//
#include <hip/hip_runtime.h>

// ---- fixed problem sizes -------------------------------------------------
#define BB 4
#define TT 1024
#define EE 1024
#define HH 16
#define DD 64
#define FF 4096          // 4*E
#define BT 4096          // B*T
#define KC 4             // k-steps (of 32) per TDM chunk: 16KB per chunk

typedef __attribute__((ext_vector_type(16))) __bf16   v16bf;
typedef __attribute__((ext_vector_type(8)))  float    v8f;
typedef __attribute__((ext_vector_type(4)))  unsigned u32x4;
typedef __attribute__((ext_vector_type(8)))  int      i32x8;
typedef __attribute__((ext_vector_type(4)))  int      i32x4;

union FragU { v16bf v; u32x4 q[2]; };
union PackU { __bf16 b[2]; unsigned u; };

#ifndef __has_builtin
#define __has_builtin(x) 0
#endif
#if __has_builtin(__builtin_amdgcn_tensor_load_to_lds) && \
    __has_builtin(__builtin_amdgcn_s_wait_tensorcnt)
#define HAVE_TDM 1
#else
#define HAVE_TDM 0
#endif

static __device__ inline v8f wmma_bf16(v16bf a, v16bf b, v8f c) {
  return __builtin_amdgcn_wmma_f32_16x16x32_bf16(false, a, false, b,
                                                 (short)0, c, false, false);
}

#if HAVE_TDM
// 1-D TDM copy: nelem bf16 elements (nelem <= 65535), global -> LDS.
// D# layout per CDNA5 ISA:
//  g0: w0 = count(1); w1 = lds_addr; w2 = global_addr[31:0];
//      w3 = global_addr[56:32] | type=2 << 30
//  g1: w0 = data_size(2B)<<16; w1[31:16] = tensor_dim0 lo16;
//      w2 = tensor_dim0 hi16 | tensor_dim1(1)<<16; w3[31:16] = tile_dim0;
//      w5 = tensor_dim0_stride lo32
// This toolchain's builtin takes 6 args (clang-23 / therock-10.0 form):
//  (uint32x4 g0, int32x8 g1, int32x4 g2, int32x4 g3, int32x8 g4, i32 cpol)
static __device__ inline void tdm_load_1d(unsigned lds_off, const void* gptr,
                                          unsigned nelem) {
  unsigned long long ga = (unsigned long long)(size_t)gptr;
  u32x4 g0 = { 1u, lds_off, (unsigned)(ga & 0xffffffffu),
               (unsigned)((ga >> 32) & 0x01ffffffu) | (2u << 30) };
  i32x8 g1 = { (int)(1u << 16),
               (int)(nelem << 16),
               (int)(((nelem >> 16) & 0xffffu) | (1u << 16)),
               (int)(nelem << 16),
               0,
               (int)nelem,
               0, 0 };
  i32x4 gz4 = { 0, 0, 0, 0 };
  i32x8 gz8 = { 0, 0, 0, 0, 0, 0, 0, 0 };
  __builtin_amdgcn_tensor_load_to_lds(g0, g1, gz4, gz4, gz8, 0);
}
#endif

// ---- weight pre-pack: fp32 -> bf16 B-fragment order ----------------------
// fragment index f = (ntg*(K/32) + kt)*4 + nl   (ntg = ntile/4, nl = ntile%4)
// -> one (4 ntile x KC k-step) chunk is 16KB contiguous for the TDM.
// within fragment: dword (lane*8 + j); lane -> col n = ntile*16 + (lane&15);
// dword j holds K = kt*32 + 2*(j&3) + 8*(lane>>4) + 16*(j>>2), +1
__global__ void pack_w(const float* __restrict__ src, unsigned* __restrict__ dst,
                       int K, int N, int qkv) {
  int idx  = blockIdx.x * 256 + threadIdx.x;
  int f    = idx >> 8;
  int rr   = idx & 255;
  int lane = rr >> 3;
  int j    = rr & 7;
  int ktn  = K >> 5;
  int nl   = f & 3;
  int kt   = (f >> 2) % ktn;
  int ntg  = (f >> 2) / ktn;
  int ncol = (ntg * 4 + nl) * 16 + (lane & 15);
  int k    = kt * 32 + 2 * (j & 3) + 8 * (lane >> 4) + 16 * (j >> 2);
  size_t s0, stride;
  if (qkv) {            // src is (H, K, D): col = h*D + d
    int hh = ncol >> 6, d = ncol & 63;
    s0 = ((size_t)hh * K + k) * DD + d;  stride = DD;
  } else {              // src is (K, N) row-major
    s0 = (size_t)k * N + ncol;           stride = (size_t)N;
  }
  PackU p;
  p.b[0] = (__bf16)src[s0];
  p.b[1] = (__bf16)src[s0 + stride];
  dst[idx] = p.u;
}

// ---- LayerNorm: fp32 row -> bf16 row (block per row, 256 thr) ------------
__global__ void ln_kernel(const float* __restrict__ x, const float* __restrict__ g,
                          const float* __restrict__ bt, __bf16* __restrict__ out) {
  int row = blockIdx.x, tid = threadIdx.x;
  int wid = tid >> 5, lane = tid & 31;
  const float* xr = x + (size_t)row * EE;
  float v[4], s = 0.f, s2 = 0.f;
#pragma unroll
  for (int k = 0; k < 4; ++k) {
    v[k] = xr[tid + k * 256];
    s += v[k]; s2 += v[k] * v[k];
  }
#pragma unroll
  for (int m = 1; m < 32; m <<= 1) {
    s  += __shfl_xor(s,  m, 32);
    s2 += __shfl_xor(s2, m, 32);
  }
  __shared__ float rs[8], rs2[8];
  if (lane == 0) { rs[wid] = s; rs2[wid] = s2; }
  __syncthreads();
  float ts = 0.f, ts2 = 0.f;
#pragma unroll
  for (int w = 0; w < 8; ++w) { ts += rs[w]; ts2 += rs2[w]; }
  float mu  = ts  * (1.0f / EE);
  float var = ts2 * (1.0f / EE) - mu * mu;
  float rst = rsqrtf(var + 1e-5f);
#pragma unroll
  for (int k = 0; k < 4; ++k) {
    int i = tid + k * 256;
    out[(size_t)row * EE + i] = (__bf16)((v[k] - mu) * rst * g[i] + bt[i]);
  }
}

// ---- WMMA GEMM: block = 128 rows x 64 cols; wave = 16 rows x 64 cols -----
// B tile streamed global->LDS by the Tensor Data Mover (wave 0), double
// buffered; all 8 waves consume the same LDS fragments (8x traffic cut).
// MODE 0: store bf16 scattered to [b,h,t,d]          (q / k)
// MODE 1: store bf16 scattered to [b,h,d,t]          (v transposed)
// MODE 2: outF = resid + acc + bias   (fp32, row-major [M,N])
// MODE 3: outB = relu(acc + bias)     (bf16, row-major [M,N])
template <int MODE>
__global__ void gemm16(const __bf16* __restrict__ A, const unsigned* __restrict__ Bp,
                       const float* __restrict__ bias, const float* __restrict__ resid,
                       float* __restrict__ outF, __bf16* __restrict__ outB,
                       int N, int K) {
  int widx = threadIdx.x >> 5, lane = threadIdx.x & 31;
  int n = lane & 15, hi = lane >> 4;
  int mtile = blockIdx.y * 8 + widx;
  int ntg   = blockIdx.x;            // 4 column tiles (64 cols) per block
  int ktn   = K >> 5;
  const unsigned* Au = (const unsigned*)A;
  size_t abase = (size_t)(mtile * 16 + n) * K / 2;

  v8f zero = {0.f, 0.f, 0.f, 0.f, 0.f, 0.f, 0.f, 0.f};
  v8f acc[4];
#pragma unroll
  for (int j = 0; j < 4; ++j) acc[j] = zero;

#if HAVE_TDM
  __shared__ unsigned Bs[2][KC * 4 * 256];   // 2 x 16KB B chunks
  const unsigned* bstrip = Bp + ((size_t)ntg * ktn) * 1024;  // 4 frags/kt * 256
  int chunks = ktn / KC;
  if (widx == 0)
    tdm_load_1d((unsigned)(size_t)(void*)&Bs[0][0], bstrip, KC * 4 * 512);
  for (int c = 0; c < chunks; ++c) {
    if (widx == 0) {
      if (c + 1 < chunks) {
        tdm_load_1d((unsigned)(size_t)(void*)&Bs[(c + 1) & 1][0],
                    bstrip + (size_t)(c + 1) * (KC * 4 * 256), KC * 4 * 512);
        __builtin_amdgcn_s_wait_tensorcnt((short)1);
      } else {
        __builtin_amdgcn_s_wait_tensorcnt((short)0);
      }
    }
    __syncthreads();                 // chunk c resident in LDS for all waves
    const unsigned* bbuf = &Bs[c & 1][0];
#pragma unroll
    for (int kl = 0; kl < KC; ++kl) {
      int kt = c * KC + kl;
      FragU a;
      a.q[0] = *(const u32x4*)(Au + abase + kt * 16 + 4 * hi);
      a.q[1] = *(const u32x4*)(Au + abase + kt * 16 + 8 + 4 * hi);
#pragma unroll
      for (int j = 0; j < 4; ++j) {
        const unsigned* bp = bbuf + ((kl * 4 + j) << 8) + lane * 8;
        FragU b;
        b.q[0] = *(const u32x4*)(bp);
        b.q[1] = *(const u32x4*)(bp + 4);
        acc[j] = wmma_bf16(a.v, b.v, acc[j]);
      }
    }
    __syncthreads();                 // all waves done with buf before reuse
  }
#else
  for (int kt = 0; kt < ktn; ++kt) {
    FragU a;
    a.q[0] = *(const u32x4*)(Au + abase + kt * 16 + 4 * hi);
    a.q[1] = *(const u32x4*)(Au + abase + kt * 16 + 8 + 4 * hi);
#pragma unroll
    for (int j = 0; j < 4; ++j) {
      const unsigned* bp = Bp + (((size_t)ntg * ktn + kt) * 4 + j) * 256 + lane * 8;
      FragU b;
      b.q[0] = *(const u32x4*)(bp);
      b.q[1] = *(const u32x4*)(bp + 4);
      acc[j] = wmma_bf16(a.v, b.v, acc[j]);
    }
  }
#endif

#pragma unroll
  for (int j = 0; j < 4; ++j) {
    int ncol = (ntg * 4 + j) * 16 + n;
#pragma unroll
    for (int r = 0; r < 8; ++r) {
      int mrow = mtile * 16 + r + 8 * hi;      // global row in [0, BT)
      float val = acc[j][r];
      if (MODE == 0) {
        int b = mrow >> 10, t = mrow & 1023;
        int hh = ncol >> 6, d = ncol & 63;
        outB[((((size_t)b * HH + hh) << 10) + t) * DD + d] = (__bf16)val;
      } else if (MODE == 1) {
        int b = mrow >> 10, t = mrow & 1023;
        int hh = ncol >> 6, d = ncol & 63;
        outB[(((size_t)b * HH + hh) * DD + d) * TT + t] = (__bf16)val;
      } else if (MODE == 2) {
        size_t o = (size_t)mrow * N + ncol;
        outF[o] = resid[o] + val + bias[ncol];
      } else {
        float z = val + bias[ncol];
        outB[(size_t)mrow * N + ncol] = (__bf16)fmaxf(z, 0.f);
      }
    }
  }
}

// ---- flash attention: 1 wave per (b,h, 16-row Q tile), keys 32/step ------
__global__ void attn_kernel(const __bf16* __restrict__ qb, const __bf16* __restrict__ kb,
                            const __bf16* __restrict__ vtb, __bf16* __restrict__ ob) {
  __shared__ __bf16 p_lds[8 * 512];   // per-wave 16x32 bf16 P tile
  int widx = threadIdx.x >> 5, lane = threadIdx.x & 31;
  int n = lane & 15, hi = lane >> 4;
  int w = blockIdx.x * 8 + widx;      // 4096 wave tasks
  int bh = w >> 6;                    // T/16 = 64 tiles per (b,h)
  int tt = w & 63;

  const unsigned* qu = (const unsigned*)qb;
  const unsigned* ku = (const unsigned*)kb;
  const unsigned* vu = (const unsigned*)vtb;

  // Q A-fragments (16 x 64, two K=32 fragments), row m = lane&15
  FragU qa[2];
  size_t qbase = ((size_t)bh * TT + tt * 16 + n) * DD / 2;
#pragma unroll
  for (int j = 0; j < 2; ++j) {
    qa[j].q[0] = *(const u32x4*)(qu + qbase + 16 * j + 8 * hi);
    qa[j].q[1] = *(const u32x4*)(qu + qbase + 16 * j + 8 * hi + 4);
  }

  v8f zero = {0.f, 0.f, 0.f, 0.f, 0.f, 0.f, 0.f, 0.f};
  v8f oacc[4];
#pragma unroll
  for (int j = 0; j < 4; ++j) oacc[j] = zero;
  float mr[8], lr[8];
#pragma unroll
  for (int r = 0; r < 8; ++r) { mr[r] = -1e30f; lr[r] = 0.f; }

  __bf16* pw = p_lds + widx * 512;
  const unsigned* plu = (const unsigned*)pw;

  for (int s0 = 0; s0 < TT; s0 += 32) {
    // scores: two 16x16 tiles over d=0..63
    v8f sc[2];
#pragma unroll
    for (int st = 0; st < 2; ++st) {
      size_t kbase = ((size_t)bh * TT + s0 + st * 16 + n) * DD / 2;
      FragU k0, k1;
      k0.q[0] = *(const u32x4*)(ku + kbase + 8 * hi);
      k0.q[1] = *(const u32x4*)(ku + kbase + 8 * hi + 4);
      k1.q[0] = *(const u32x4*)(ku + kbase + 16 + 8 * hi);
      k1.q[1] = *(const u32x4*)(ku + kbase + 16 + 8 * hi + 4);
      v8f c = wmma_bf16(qa[0].v, k0.v, zero);
      sc[st] = wmma_bf16(qa[1].v, k1.v, c);
    }
    // online softmax (rows r+8*hi, cols striped across 16-lane halves)
#pragma unroll
    for (int r = 0; r < 8; ++r) {
      float a0 = sc[0][r], a1 = sc[1][r];
      float cm = fmaxf(a0, a1);
#pragma unroll
      for (int m = 1; m < 16; m <<= 1) cm = fmaxf(cm, __shfl_xor(cm, m, 32));
      float nm   = fmaxf(mr[r], cm);
      float corr = __expf(mr[r] - nm);
      float p0   = __expf(a0 - nm);
      float p1   = __expf(a1 - nm);
      float ls   = p0 + p1;
#pragma unroll
      for (int m = 1; m < 16; m <<= 1) ls += __shfl_xor(ls, m, 32);
      lr[r] = lr[r] * corr + ls;
      mr[r] = nm;
#pragma unroll
      for (int j = 0; j < 4; ++j) oacc[j][r] *= corr;
      int mm = r + 8 * hi;
      pw[mm * 32 + n]      = (__bf16)p0;
      pw[mm * 32 + 16 + n] = (__bf16)p1;
    }
    __syncthreads();   // order bf16 P stores before b128 P loads
    FragU pa;          // P as A-fragment: 16 rows x K=32
    pa.q[0] = *(const u32x4*)(plu + n * 16 + 8 * hi);
    pa.q[1] = *(const u32x4*)(plu + n * 16 + 8 * hi + 4);
    // O += P * V   (V transposed [b,h,d,t] -> contiguous s-pairs)
#pragma unroll
    for (int dt = 0; dt < 4; ++dt) {
      size_t vbase = ((size_t)bh * DD + dt * 16 + n) * TT / 2 + s0 / 2;
      FragU vf;
      vf.q[0] = *(const u32x4*)(vu + vbase + 8 * hi);
      vf.q[1] = *(const u32x4*)(vu + vbase + 8 * hi + 4);
      oacc[dt] = wmma_bf16(pa.v, vf.v, oacc[dt]);
    }
    __syncthreads();   // keep next-iteration P stores after this iter's loads
  }
  // normalize + scatter o to row-major [b*T+t, h*D+d]
  int b = bh >> 4, h = bh & 15;
#pragma unroll
  for (int r = 0; r < 8; ++r) {
    float inv = 1.f / lr[r];
    int t = tt * 16 + r + 8 * hi;
#pragma unroll
    for (int dt = 0; dt < 4; ++dt) {
      int d = dt * 16 + n;
      ob[((size_t)b * TT + t) * EE + h * DD + d] = (__bf16)(oacc[dt][r] * inv);
    }
  }
}

// --------------------------------------------------------------------------
extern "C" void kernel_launch(void* const* d_in, const int* in_sizes, int n_in,
                              void* d_out, int out_size, void* d_ws, size_t ws_size,
                              hipStream_t stream) {
  const float* x     = (const float*)d_in[0];
  const float* ln1_g = (const float*)d_in[1];
  const float* ln1_b = (const float*)d_in[2];
  const float* Wq    = (const float*)d_in[3];
  const float* Wk    = (const float*)d_in[4];
  const float* Wv    = (const float*)d_in[5];
  const float* Wp    = (const float*)d_in[6];
  const float* bp    = (const float*)d_in[7];
  const float* ln2_g = (const float*)d_in[8];
  const float* ln2_b = (const float*)d_in[9];
  const float* W1    = (const float*)d_in[10];
  const float* b1    = (const float*)d_in[11];
  const float* W2    = (const float*)d_in[12];
  const float* b2    = (const float*)d_in[13];
  float* out = (float*)d_out;

  char* ws = (char*)d_ws;
  size_t off = 0;
  auto alloc = [&](size_t bytes) -> char* {
    char* p = ws + off;
    off = (off + bytes + 255) & ~(size_t)255;
    return p;
  };
  __bf16*   h_bf  = (__bf16*)alloc((size_t)BT * EE * 2);
  __bf16*   qbuf  = (__bf16*)alloc((size_t)BT * EE * 2);
  __bf16*   kbuf  = (__bf16*)alloc((size_t)BT * EE * 2);
  __bf16*   vtbuf = (__bf16*)alloc((size_t)BT * EE * 2);
  __bf16*   obuf  = (__bf16*)alloc((size_t)BT * EE * 2);
  float*    res1  = (float*) alloc((size_t)BT * EE * 4);
  __bf16*   h2_bf = (__bf16*)alloc((size_t)BT * EE * 2);
  __bf16*   a1buf = (__bf16*)alloc((size_t)BT * FF * 2);
  unsigned* wq_p  = (unsigned*)alloc((size_t)EE * EE * 2);
  unsigned* wk_p  = (unsigned*)alloc((size_t)EE * EE * 2);
  unsigned* wv_p  = (unsigned*)alloc((size_t)EE * EE * 2);
  unsigned* wp_p  = (unsigned*)alloc((size_t)EE * EE * 2);
  unsigned* w1_p  = (unsigned*)alloc((size_t)EE * FF * 2);
  unsigned* w2_p  = (unsigned*)alloc((size_t)FF * EE * 2);

  // 1) pack weights (fp32 -> bf16 fragment order, TDM-chunk contiguous)
  pack_w<<<EE * EE / 512, 256, 0, stream>>>(Wq, wq_p, EE, EE, 1);
  pack_w<<<EE * EE / 512, 256, 0, stream>>>(Wk, wk_p, EE, EE, 1);
  pack_w<<<EE * EE / 512, 256, 0, stream>>>(Wv, wv_p, EE, EE, 1);
  pack_w<<<EE * EE / 512, 256, 0, stream>>>(Wp, wp_p, EE, EE, 0);
  pack_w<<<EE * FF / 512, 256, 0, stream>>>(W1, w1_p, EE, FF, 0);
  pack_w<<<FF * EE / 512, 256, 0, stream>>>(W2, w2_p, FF, EE, 0);

  // 2) LN1
  ln_kernel<<<BT, 256, 0, stream>>>(x, ln1_g, ln1_b, h_bf);

  // 3) QKV projections
  gemm16<0><<<dim3(EE / 64, BT / 128), 256, 0, stream>>>(h_bf, wq_p, nullptr, nullptr, nullptr, qbuf,  EE, EE);
  gemm16<0><<<dim3(EE / 64, BT / 128), 256, 0, stream>>>(h_bf, wk_p, nullptr, nullptr, nullptr, kbuf,  EE, EE);
  gemm16<1><<<dim3(EE / 64, BT / 128), 256, 0, stream>>>(h_bf, wv_p, nullptr, nullptr, nullptr, vtbuf, EE, EE);

  // 4) flash attention (non-causal, no 1/sqrt(D) scaling, per reference)
  attn_kernel<<<BB * HH * (TT / 16) / 8, 256, 0, stream>>>(qbuf, kbuf, vtbuf, obuf);

  // 5) output projection + residual -> res1 (fp32)
  gemm16<2><<<dim3(EE / 64, BT / 128), 256, 0, stream>>>(obuf, wp_p, bp, x, res1, nullptr, EE, EE);

  // 6) LN2
  ln_kernel<<<BT, 256, 0, stream>>>(res1, ln2_g, ln2_b, h2_bf);

  // 7) MLP
  gemm16<3><<<dim3(FF / 64, BT / 128), 256, 0, stream>>>(h2_bf, w1_p, b1, nullptr, nullptr, a1buf, FF, EE);
  gemm16<2><<<dim3(EE / 64, BT / 128), 256, 0, stream>>>(a1buf, w2_p, b2, res1, out, nullptr, EE, FF);
}
